// TransformerEncoder_83150566851084
// MI455X (gfx1250) — matrix operand order repaired and checked by reading, not compiled, not run on hardware
//
#include <hip/hip_runtime.h>
#include <math.h>

// ---------------- problem constants ----------------
#define NB     8
#define SEQ    1024
#define DMODEL 256
#define NHEAD  4
#define DHEAD  64
#define FFDIM  1024
#define NLAYER 2
#define NCLS   10

typedef __attribute__((ext_vector_type(16))) __bf16 v16bf;
typedef __attribute__((ext_vector_type(8)))  __bf16 v8bf;
typedef __attribute__((ext_vector_type(2)))  __bf16 v2bf;
typedef __attribute__((ext_vector_type(8)))  float  v8f;

// ---------------- bf16 conversion helpers ----------------
#if __has_builtin(__builtin_amdgcn_cvt_pk_bf16_f32)
__device__ __forceinline__ v2bf pkbf(float a, float b) {
  return __builtin_amdgcn_cvt_pk_bf16_f32(a, b);
}
#else
__device__ __forceinline__ v2bf pkbf(float a, float b) {
  v2bf r; r[0] = (__bf16)a; r[1] = (__bf16)b; return r;
}
#endif
__device__ __forceinline__ float bf2f(__bf16 b) { return (float)b; }

__device__ __forceinline__ v16bf cat8(v8bf lo, v8bf hi) {
  return __builtin_shufflevector(lo, hi, 0,1,2,3,4,5,6,7,8,9,10,11,12,13,14,15);
}

// ---------------- CDNA5 async global -> LDS copy (16B per lane) ----------------
__device__ __forceinline__ void async_cp16(void* lds, const void* g) {
  unsigned off = (unsigned)(uintptr_t)lds;     // flat LDS addr: low 32 bits = LDS offset
  asm volatile("global_load_async_to_lds_b128 %0, %1, off"
               :: "v"(off), "v"(g) : "memory");
}
__device__ __forceinline__ void wait_async0() {
  asm volatile("s_wait_asynccnt 0x0" ::: "memory");
}
__device__ __forceinline__ void wait_async2() {   // allow 2 newest copies in flight
  asm volatile("s_wait_asynccnt 0x2" ::: "memory");
}

// =====================================================================
// embed + positional encoding: h = emb[x] + pe  (fp32 + bf16 mirror)
// =====================================================================
__global__ __launch_bounds__(128)
void embed_kernel(const int* __restrict__ x, const float* __restrict__ emb,
                  const float* __restrict__ pe, float* __restrict__ h,
                  __bf16* __restrict__ hbf) {
  int row = blockIdx.x;               // b*SEQ + s
  int d   = threadIdx.x * 2;
  int s   = row & (SEQ - 1);
  int id  = x[row];
  float2 e = *(const float2*)&emb[(size_t)id * DMODEL + d];
  float2 p = *(const float2*)&pe[s * DMODEL + d];
  float v0 = e.x + p.x, v1 = e.y + p.y;
  *(float2*)&h[(size_t)row * DMODEL + d] = float2{v0, v1};
  *(v2bf*)&hbf[(size_t)row * DMODEL + d] = pkbf(v0, v1);
}

// =====================================================================
// one-time weight convert+transpose: src[K][N] fp32 -> dst[N][K] bf16
// =====================================================================
__global__ __launch_bounds__(256)
void transpose_bf16_kernel(const float* __restrict__ src, __bf16* __restrict__ dst,
                           int K, int N) {
  int id = blockIdx.x * 256 + threadIdx.x;
  if (id >= K * N) return;
  int n = id % N, k = id / N;          // consecutive id -> consecutive n (coalesced reads)
  dst[(size_t)n * K + k] = pkbf(src[(size_t)k * N + n], 0.f)[0];
}

// =====================================================================
// GEMM: C = act(A @ W + bias); A[M][K] bf16 row-major, Wt[N][K] bf16.
// Double-buffered async global->LDS staging overlapped with WMMA.
// C (fp32) and Cbf (bf16) outputs each optional.
// =====================================================================
__global__ __launch_bounds__(128)
void gemm_bias_kernel(const __bf16* __restrict__ A, const __bf16* __restrict__ Wt,
                      const float* __restrict__ bias, float* __restrict__ C,
                      __bf16* __restrict__ Cbf, int M, int N, int K, int relu) {
  __shared__ __attribute__((aligned(16))) __bf16 As[2][64][40];
  __shared__ __attribute__((aligned(16))) __bf16 Bt[2][64][40];
  const int tid  = threadIdx.x;
  const int wv   = tid >> 5, lane = tid & 31;
  const int hh   = lane >> 4, mn = lane & 15;
  const int m0   = blockIdx.y * 64, n0 = blockIdx.x * 64;

  // per-thread staging assignment: 16B chunk (srow, sq) and (srow+32, sq)
  const int srow = tid >> 2, sq = (tid & 3) * 8;
  const __bf16* aSrc0 = A  + (size_t)(m0 + srow)      * K + sq;
  const __bf16* aSrc1 = A  + (size_t)(m0 + srow + 32) * K + sq;
  const __bf16* bSrc0 = Wt + (size_t)(n0 + srow)      * K + sq;
  const __bf16* bSrc1 = Wt + (size_t)(n0 + srow + 32) * K + sq;

  auto stage = [&](int buf, int k0) {
    async_cp16(&As[buf][srow][sq],      aSrc0 + k0);
    async_cp16(&As[buf][srow + 32][sq], aSrc1 + k0);
    async_cp16(&Bt[buf][srow][sq],      bSrc0 + k0);
    async_cp16(&Bt[buf][srow + 32][sq], bSrc1 + k0);
  };

  v8f acc[4] = {v8f{}, v8f{}, v8f{}, v8f{}};

  stage(0, 0);
  wait_async0();
  __syncthreads();

  int cur = 0;
  const int arow = wv * 16 + mn;
  for (int k0 = 0; k0 < K; k0 += 32) {
    if (k0 + 32 < K) stage(cur ^ 1, k0 + 32);   // overlap next tile with WMMA

    v16bf a = cat8(*(const v8bf*)&As[cur][arow][8 * hh],
                   *(const v8bf*)&As[cur][arow][16 + 8 * hh]);
    #pragma unroll
    for (int ns = 0; ns < 4; ++ns) {
      v16bf b = cat8(*(const v8bf*)&Bt[cur][ns * 16 + mn][8 * hh],
                     *(const v8bf*)&Bt[cur][ns * 16 + mn][16 + 8 * hh]);
      acc[ns] = __builtin_amdgcn_wmma_f32_16x16x32_bf16(
          false, a, false, b, (short)0, acc[ns], false, false);
    }
    wait_async0();
    __syncthreads();
    cur ^= 1;
  }

  #pragma unroll
  for (int ns = 0; ns < 4; ++ns) {
    #pragma unroll
    for (int v = 0; v < 8; ++v) {
      int row = wv * 16 + v + 8 * hh;
      int col = n0 + ns * 16 + mn;
      float val = acc[ns][v] + bias[col];
      if (relu) val = fmaxf(val, 0.f);
      if (C)   C[(size_t)(m0 + row) * N + col] = val;
      if (Cbf) Cbf[(size_t)(m0 + row) * N + col] = pkbf(val, val)[0];
    }
  }
}

// =====================================================================
// fused attention per (bh, 16-row q-tile), bf16 q/k/v inputs:
// scores -> ==0 hack -> /16 -> softmax (bf16 strip in LDS) -> PV.
// K tiles: wave-private async double-buffer with partial asynccnt waits.
// =====================================================================
__global__ __launch_bounds__(128)
void attention_kernel(const __bf16* __restrict__ Qb, const __bf16* __restrict__ Kb,
                      const __bf16* __restrict__ Vb, const int* __restrict__ lengths,
                      float* __restrict__ ctx) {
  __shared__ __attribute__((aligned(16))) __bf16 sc[16][SEQ + 8];     // probs strip
  __shared__ __attribute__((aligned(16))) __bf16 Qt[16][72];          // Q tile [m][k]
  __shared__ __attribute__((aligned(16))) __bf16 Kt[4][2][16][72];    // per-wave K tiles (pingpong)
  __shared__ __attribute__((aligned(16))) __bf16 Vtt[2][64][40];      // V tiles transposed (pingpong)
  __shared__ float red[16][8];
  __shared__ float rmax[16], rsum[16];

  const int tid  = threadIdx.x;
  const int wv   = tid >> 5, lane = tid & 31;
  const int hh   = lane >> 4, mn = lane & 15;
  const int qt   = blockIdx.x;            // 0..63
  const int bh   = blockIdx.y;            // 0..31 (b*NHEAD+head)
  const int b    = bh >> 2, head = bh & 3;
  // faithful bug: only first NB entries of the flattened B*H dim are masked
  const int len  = (bh < NB) ? lengths[bh] : SEQ;

  const __bf16* Qp = Qb + (size_t)b * SEQ * DMODEL + head * DHEAD;
  const __bf16* Kp = Kb + (size_t)b * SEQ * DMODEL + head * DHEAD;
  const __bf16* Vp = Vb + (size_t)b * SEQ * DMODEL + head * DHEAD;

  // K staging assignment: lane covers chunks (kr, kq) and (kr+8, kq)
  const int kr = lane >> 2, kq = (lane & 3) * 8;
  const __bf16* kSrc0 = Kp + (size_t)kr * DMODEL + kq;
  const __bf16* kSrc1 = Kp + (size_t)(kr + 8) * DMODEL + kq;
  auto stage_k = [&](int buf, int kt) {
    size_t off = (size_t)kt * 16 * DMODEL;
    async_cp16(&Kt[wv][buf][kr][kq],     kSrc0 + off);
    async_cp16(&Kt[wv][buf][kr + 8][kq], kSrc1 + off);
  };
  auto mask_k = [&](int buf, int kt) {
    if (kt * 16 + kr     >= len) { v8bf z = {}; *(v8bf*)&Kt[wv][buf][kr][kq]     = z; }
    if (kt * 16 + kr + 8 >= len) { v8bf z = {}; *(v8bf*)&Kt[wv][buf][kr + 8][kq] = z; }
    asm volatile("s_wait_dscnt 0x0" ::: "memory");
  };

  // ---- async stage Q tile, then first K tile; drain Q with partial wait ----
  const int qr = tid >> 3, qq = (tid & 7) * 8;
  const int qs = qt * 16 + qr;
  async_cp16(&Qt[qr][qq], Qp + (size_t)qs * DMODEL + qq);
  stage_k(0, wv);
  wait_async2();                           // Q drained; K tile 0 may be in flight
  if (qs >= len) { v8bf z = {}; *(v8bf*)&Qt[qr][qq] = z; }
  __syncthreads();

  // Q fragments (dh=64 -> two 32-wide K steps), hoisted
  v16bf aq[2];
  #pragma unroll
  for (int ks = 0; ks < 2; ++ks)
    aq[ks] = cat8(*(const v8bf*)&Qt[mn][ks * 32 + 8 * hh],
                  *(const v8bf*)&Qt[mn][ks * 32 + 16 + 8 * hh]);

  // ---- scores = Q @ K^T; wave-private double-buffered K tiles ----
  {
    int cur = 0;
    for (int kt = wv; kt < SEQ / 16; kt += 4) {
      if (kt + 4 < SEQ / 16) { stage_k(cur ^ 1, kt + 4); wait_async2(); }
      else                   { wait_async0(); }
      mask_k(cur, kt);

      v8f acc = {};
      #pragma unroll
      for (int ks = 0; ks < 2; ++ks) {
        v16bf bk = cat8(*(const v8bf*)&Kt[wv][cur][mn][ks * 32 + 8 * hh],
                        *(const v8bf*)&Kt[wv][cur][mn][ks * 32 + 16 + 8 * hh]);
        acc = __builtin_amdgcn_wmma_f32_16x16x32_bf16(
            false, aq[ks], false, bk, (short)0, acc, false, false);
      }
      #pragma unroll
      for (int v = 0; v < 8; ++v) {
        int row = v + 8 * hh;
        float s = acc[v];
        s = (s == 0.f) ? 1e-10f : s;       // faithful ==0 hack (pre-scale)
        s *= 0.0625f;                      // / sqrt(D) = /16
        sc[row][kt * 16 + mn] = pkbf(s, s)[0];
      }
      cur ^= 1;
    }
  }
  __syncthreads();

  // ---- softmax over the 1024-wide strip (store unnormalized exp) ----
  {
    const int r = tid & 15, seg = tid >> 4;    // 8 threads x 128 cols per row
    float mx = -3.4e38f;
    for (int c = seg * 128; c < seg * 128 + 128; ++c)
      mx = fmaxf(mx, bf2f(sc[r][c]));
    red[r][seg] = mx;
    __syncthreads();
    if (seg == 0) {
      float m2 = red[r][0];
      #pragma unroll
      for (int j = 1; j < 8; ++j) m2 = fmaxf(m2, red[r][j]);
      rmax[r] = m2;
    }
    __syncthreads();
    float m2 = rmax[r], sum = 0.f;
    for (int c = seg * 128; c < seg * 128 + 128; c += 2) {
      float e0 = __expf(bf2f(sc[r][c])     - m2);
      float e1 = __expf(bf2f(sc[r][c + 1]) - m2);
      *(v2bf*)&sc[r][c] = pkbf(e0, e1);
      sum += e0 + e1;
    }
    red[r][seg] = sum;
    __syncthreads();
    if (seg == 0) {
      float s2 = 0.f;
      #pragma unroll
      for (int j = 0; j < 8; ++j) s2 += red[r][j];
      rsum[r] = s2;
    }
    __syncthreads();
  }

  // ---- ctx = P @ V; pingpong V tiles, one barrier per step ----
  {
    const int vc = tid & 63, vkh = tid >> 6;   // staging: col vc, k-half vkh
    auto stage_v = [&](int buf, int kk) {
      #pragma unroll
      for (int j = 0; j < 16; j += 2) {
        int s0 = kk * 32 + vkh * 16 + j;
        v2bf p = {};
        if (s0     < len) p[0] = Vp[(size_t)s0       * DMODEL + vc];
        if (s0 + 1 < len) p[1] = Vp[(size_t)(s0 + 1) * DMODEL + vc];
        *(v2bf*)&Vtt[buf][vc][vkh * 16 + j] = p;
      }
    };

    stage_v(0, 0);
    int cur = 0;
    v8f acc = {};
    for (int kk = 0; kk < SEQ / 32; ++kk) {
      __syncthreads();                     // staged 'cur' visible; old reads done
      if (kk + 1 < SEQ / 32) stage_v(cur ^ 1, kk + 1);
      v16bf a  = cat8(*(const v8bf*)&sc[mn][kk * 32 + 8 * hh],
                      *(const v8bf*)&sc[mn][kk * 32 + 16 + 8 * hh]);
      v16bf bv = cat8(*(const v8bf*)&Vtt[cur][wv * 16 + mn][8 * hh],
                      *(const v8bf*)&Vtt[cur][wv * 16 + mn][16 + 8 * hh]);
      acc = __builtin_amdgcn_wmma_f32_16x16x32_bf16(
          false, a, false, bv, (short)0, acc, false, false);
      cur ^= 1;
    }
    float* Cp = ctx + (size_t)b * SEQ * DMODEL + head * DHEAD;
    #pragma unroll
    for (int v = 0; v < 8; ++v) {
      int row = v + 8 * hh;
      int qrow = qt * 16 + row;
      Cp[(size_t)qrow * DMODEL + wv * 16 + mn] = acc[v] / rsum[row];
    }
  }
}

// =====================================================================
// out = LayerNorm(a + b) * g + beta  (+ bf16 mirror)
// =====================================================================
__global__ __launch_bounds__(DMODEL)
void add_ln_kernel(const float* __restrict__ a, const float* __restrict__ bb,
                   const float* __restrict__ g, const float* __restrict__ beta,
                   float* __restrict__ out, __bf16* __restrict__ outbf) {
  __shared__ float sd[DMODEL];
  int row = blockIdx.x, t = threadIdx.x;
  size_t idx = (size_t)row * DMODEL + t;
  float x = a[idx] + bb[idx];
  sd[t] = x; __syncthreads();
  for (int s = DMODEL / 2; s > 0; s >>= 1) { if (t < s) sd[t] += sd[t + s]; __syncthreads(); }
  float mu = sd[0] * (1.f / DMODEL);
  __syncthreads();
  float dx = x - mu;
  sd[t] = dx * dx; __syncthreads();
  for (int s = DMODEL / 2; s > 0; s >>= 1) { if (t < s) sd[t] += sd[t + s]; __syncthreads(); }
  float var = sd[0] * (1.f / DMODEL);
  float y = dx * rsqrtf(var + 1e-5f) * g[t] + beta[t];
  out[idx]   = y;
  outbf[idx] = pkbf(y, y)[0];
}

// =====================================================================
// masked mean pool: pooled[b,d] = sum_{s<len} h[b,s,d] / SEQ
// =====================================================================
__global__ __launch_bounds__(DMODEL)
void pool_kernel(const float* __restrict__ h, const int* __restrict__ lengths,
                 float* __restrict__ pooled) {
  int b = blockIdx.x, d = threadIdx.x;
  int len = lengths[b];
  float s = 0.f;
  for (int i = 0; i < len; ++i) s += h[((size_t)b * SEQ + i) * DMODEL + d];
  pooled[b * DMODEL + d] = s * (1.f / SEQ);
}

// =====================================================================
// classifier: softmax((pooled@Wc1+bc1)@Wc2+bc2)   (tiny; one block)
// =====================================================================
__global__ __launch_bounds__(256)
void classifier_kernel(const float* __restrict__ pooled, const float* __restrict__ Wc1,
                       const float* __restrict__ bc1, const float* __restrict__ Wc2,
                       const float* __restrict__ bc2, float* __restrict__ out) {
  __shared__ float hid[NB][FFDIM];
  __shared__ float lg[NB][NCLS];
  int t = threadIdx.x;
  for (int i = t; i < NB * FFDIM; i += 256) {
    int b = i >> 10, f = i & (FFDIM - 1);
    float s = bc1[f];
    for (int k = 0; k < DMODEL; ++k) s += pooled[b * DMODEL + k] * Wc1[(size_t)k * FFDIM + f];
    hid[b][f] = s;
  }
  __syncthreads();
  if (t < NB * NCLS) {
    int b = t / NCLS, c = t % NCLS;
    float s = bc2[c];
    for (int k = 0; k < FFDIM; ++k) s += hid[b][k] * Wc2[k * NCLS + c];
    lg[b][c] = s;
  }
  __syncthreads();
  if (t < NB) {
    float mx = lg[t][0];
    #pragma unroll
    for (int c = 1; c < NCLS; ++c) mx = fmaxf(mx, lg[t][c]);
    float ex[NCLS]; float sum = 0.f;
    #pragma unroll
    for (int c = 0; c < NCLS; ++c) { ex[c] = __expf(lg[t][c] - mx); sum += ex[c]; }
    #pragma unroll
    for (int c = 0; c < NCLS; ++c) out[t * NCLS + c] = ex[c] / sum;
  }
}

// =====================================================================
extern "C" void kernel_launch(void* const* d_in, const int* in_sizes, int n_in,
                              void* d_out, int out_size, void* d_ws, size_t ws_size,
                              hipStream_t stream) {
  (void)in_sizes; (void)n_in; (void)out_size; (void)ws_size;
  const int*   x       = (const int*)d_in[0];
  const int*   lengths = (const int*)d_in[1];
  const float* emb     = (const float*)d_in[2];
  const float* pe      = (const float*)d_in[3];
  const float* Wq      = (const float*)d_in[4];
  const float* bq      = (const float*)d_in[5];
  const float* Wk      = (const float*)d_in[6];
  const float* bk      = (const float*)d_in[7];
  const float* Wv      = (const float*)d_in[8];
  const float* bv      = (const float*)d_in[9];
  const float* g1      = (const float*)d_in[10];
  const float* be1     = (const float*)d_in[11];
  const float* W1      = (const float*)d_in[12];
  const float* b1      = (const float*)d_in[13];
  const float* W2      = (const float*)d_in[14];
  const float* b2      = (const float*)d_in[15];
  const float* g2      = (const float*)d_in[16];
  const float* be2     = (const float*)d_in[17];
  const float* Wc1     = (const float*)d_in[18];
  const float* bc1     = (const float*)d_in[19];
  const float* Wc2     = (const float*)d_in[20];
  const float* bc2     = (const float*)d_in[21];

  const size_t HSZ = (size_t)NB * SEQ * DMODEL;   // 2,097,152 elements
  float* f32ws  = (float*)d_ws;
  float* h      = f32ws;                 // fp32 activations
  float* ctx    = f32ws + 1 * HSZ;
  float* h1     = f32ws + 2 * HSZ;
  float* f2     = f32ws + 3 * HSZ;
  float* pooled = f32ws + 4 * HSZ;       // 2048 floats
  __bf16* bfws  = (__bf16*)(f32ws + 4 * HSZ + 4096);
  __bf16* hbf   = bfws;                  // bf16 mirrors
  __bf16* qbf   = bfws + 1 * HSZ;
  __bf16* kbf   = bfws + 2 * HSZ;
  __bf16* vbf   = bfws + 3 * HSZ;
  __bf16* h1bf  = bfws + 4 * HSZ;
  __bf16* f1bf  = bfws + 5 * HSZ;        // 4*HSZ elements
  __bf16* wt    = bfws + 9 * HSZ;        // transposed bf16 weights arena
  const size_t WQ = (size_t)DMODEL * DMODEL;   // 65536 / layer
  const size_t WF = (size_t)DMODEL * FFDIM;    // 262144 / layer
  __bf16* wqt = wt;
  __bf16* wkt = wt + 2 * WQ;
  __bf16* wvt = wt + 4 * WQ;
  __bf16* w1t = wt + 6 * WQ;             // 2*WF
  __bf16* w2t = wt + 6 * WQ + 2 * WF;    // 2*WF

  const int M = NB * SEQ;

  // one-time weight convert + transpose into [N][K] bf16
  for (int l = 0; l < NLAYER; ++l) {
    transpose_bf16_kernel<<<(int)((WQ + 255) / 256), 256, 0, stream>>>(
        Wq + l * WQ, wqt + l * WQ, DMODEL, DMODEL);
    transpose_bf16_kernel<<<(int)((WQ + 255) / 256), 256, 0, stream>>>(
        Wk + l * WQ, wkt + l * WQ, DMODEL, DMODEL);
    transpose_bf16_kernel<<<(int)((WQ + 255) / 256), 256, 0, stream>>>(
        Wv + l * WQ, wvt + l * WQ, DMODEL, DMODEL);
    transpose_bf16_kernel<<<(int)((WF + 255) / 256), 256, 0, stream>>>(
        W1 + l * WF, w1t + l * WF, DMODEL, FFDIM);
    transpose_bf16_kernel<<<(int)((WF + 255) / 256), 256, 0, stream>>>(
        W2 + l * WF, w2t + l * WF, FFDIM, DMODEL);
  }

  embed_kernel<<<M, DMODEL / 2, 0, stream>>>(x, emb, pe, h, hbf);

  for (int l = 0; l < NLAYER; ++l) {
    gemm_bias_kernel<<<dim3(DMODEL / 64, M / 64), 128, 0, stream>>>(
        hbf, wqt + l * WQ, bq + l * DMODEL, nullptr, qbf, M, DMODEL, DMODEL, 0);
    gemm_bias_kernel<<<dim3(DMODEL / 64, M / 64), 128, 0, stream>>>(
        hbf, wkt + l * WQ, bk + l * DMODEL, nullptr, kbf, M, DMODEL, DMODEL, 0);
    gemm_bias_kernel<<<dim3(DMODEL / 64, M / 64), 128, 0, stream>>>(
        hbf, wvt + l * WQ, bv + l * DMODEL, nullptr, vbf, M, DMODEL, DMODEL, 0);

    attention_kernel<<<dim3(SEQ / 16, NB * NHEAD), 128, 0, stream>>>(
        qbf, kbf, vbf, lengths, ctx);

    add_ln_kernel<<<M, DMODEL, 0, stream>>>(
        ctx, h, g1 + l * DMODEL, be1 + l * DMODEL, h1, h1bf);

    gemm_bias_kernel<<<dim3(FFDIM / 64, M / 64), 128, 0, stream>>>(
        h1bf, w1t + l * WF, b1 + l * FFDIM, nullptr, f1bf, M, FFDIM, DMODEL, 1);
    gemm_bias_kernel<<<dim3(DMODEL / 64, M / 64), 128, 0, stream>>>(
        f1bf, w2t + l * WF, b2 + l * DMODEL, f2, nullptr, M, DMODEL, FFDIM, 0);

    add_ln_kernel<<<M, DMODEL, 0, stream>>>(
        f2, h1, g2 + l * DMODEL, be2 + l * DMODEL, h, hbf);
  }

  pool_kernel<<<NB, DMODEL, 0, stream>>>(h, lengths, pooled);
  classifier_kernel<<<1, 256, 0, stream>>>(pooled, Wc1, bc1, Wc2, bc2, (float*)d_out);
}
